// DecoderRNN_71390946394833
// MI455X (gfx1250) — matrix operand order repaired and checked by reading, not compiled, not run on hardware
//
#include <hip/hip_runtime.h>
#include <math.h>

// Sizes for this problem instance
#define Bz 16
#define T1 65
#define Tz 64
#define Sz 128
#define Hz 512
#define Vz 32000
#define G3 1536   // 3*H
#define H2 1024   // 2*H
#define Mz 1024   // B*T rows

typedef __attribute__((ext_vector_type(16))) __bf16 v16bf;
typedef __attribute__((ext_vector_type(8)))  float  v8f;

union Frag {
    v16bf bf;
    unsigned int u[8];
    unsigned short us[16];
};

__device__ __forceinline__ unsigned short f2bf(float f) {
    unsigned int x = __float_as_uint(f);
    unsigned int r = x + 0x7FFFu + ((x >> 16) & 1u);   // round-to-nearest-even
    return (unsigned short)(r >> 16);
}

__device__ __forceinline__ v8f wmma_bf(const Frag& a, const Frag& b, v8f c) {
    return __builtin_amdgcn_wmma_f32_16x16x32_bf16(
        false, a.bf, false, b.bf, (short)0, c, false, false);
}

// A-fragment: 16x32 bf16 tile, source row-major (row stride = ld ushorts).
// Lane layout (ISA 7.12.2): lane<16 -> M=lane, K={0..7,16..23}; lane>=16 -> K={8..15,24..31}
__device__ __forceinline__ Frag load_a_rm(const unsigned short* p, int ld) {
    int lane = threadIdx.x & 31;
    int m  = lane & 15;
    int kb = (lane >> 4) * 8;
    Frag f;
    const unsigned short* row = p + m * ld;
#pragma unroll
    for (int v = 0; v < 8; ++v) {
        int k = ((v >> 2) * 16) + kb + 2 * (v & 3);
        f.u[v] = *(const unsigned int*)(row + k);
    }
    return f;
}

// B-fragment: 32x16, where B[k][n] = W[n0+n][k0+k] and W is row-major (ld = K stride).
// Lane layout: lane<16 -> N=lane, K={0..7,16..23}; lane>=16 -> N=lane-16, K={8..15,24..31}
__device__ __forceinline__ Frag load_b_wT(const unsigned short* p, int ld) {
    int lane = threadIdx.x & 31;
    int n  = lane & 15;
    int kb = (lane >> 4) * 8;
    Frag f;
    const unsigned short* row = p + n * ld;
#pragma unroll
    for (int v = 0; v < 8; ++v)
        f.u[v] = *(const unsigned int*)(row + ((v >> 2) * 16) + kb + 2 * (v & 3));
    return f;
}

// ---------------- prep kernels ----------------

__global__ void k_f2bf(const float* __restrict__ src, unsigned short* __restrict__ dst, int n) {
    int i = blockIdx.x * blockDim.x + threadIdx.x;
    if (i < n) dst[i] = f2bf(src[i]);
}

// xbf[(t*16+b)*H + h] = bf16(emb[inputs[b*T1+t]][h]),  t in [0,64)
__global__ void k_gather_x(const int* __restrict__ inputs, const float* __restrict__ emb,
                           unsigned short* __restrict__ xbf) {
    int idx = blockIdx.x * blockDim.x + threadIdx.x;       // Mz*Hz threads
    if (idx >= Mz * Hz) return;
    int m = idx >> 9, h = idx & (Hz - 1);
    int t = m >> 4, b = m & 15;
    int tok = inputs[b * T1 + t];
    xbf[idx] = f2bf(emb[(size_t)tok * Hz + h]);
}

// ---------------- xg GEMM: (1024x512) x (512x1536) ----------------
__global__ void k_gemm_xg(const unsigned short* __restrict__ xbf,
                          const unsigned short* __restrict__ wihbf,
                          const float* __restrict__ b_ih,
                          float* __restrict__ xg) {
    int gw = (blockIdx.x * blockDim.x + threadIdx.x) >> 5;   // 6144 waves: 64 x 96 tiles
    int mt = gw / 96, nt = gw % 96;
    int lane = threadIdx.x & 31;
    v8f c = {};
#pragma unroll 4
    for (int kk = 0; kk < 16; ++kk) {
        Frag a = load_a_rm(xbf + (size_t)mt * 16 * Hz + kk * 32, Hz);
        Frag b = load_b_wT(wihbf + (size_t)nt * 16 * Hz + kk * 32, Hz);
        c = wmma_bf(a, b, c);
    }
    int nl = lane & 15, mh = (lane >> 4) * 8;
    int n = nt * 16 + nl;
    float bias = b_ih[n];
#pragma unroll
    for (int r = 0; r < 8; ++r) {
        int m = mt * 16 + mh + r;
        xg[(size_t)m * G3 + n] = c[r] + bias;
    }
}

// ---------------- GRU scan: single WGP, LDS-resident state ----------------
__global__ __launch_bounds__(1024, 1)
void k_gru(const float* __restrict__ enc_hidden,
           const unsigned short* __restrict__ whhbf,
           const float* __restrict__ b_hh,
           const float* __restrict__ xg,
           unsigned short* __restrict__ outbf,
           float* __restrict__ hlast_out) {
    __shared__ float          sh_h[Bz][Hz];      // 32 KB
    __shared__ unsigned short sh_hbf[Bz][Hz];    // 16 KB
    __shared__ float          sh_hg[Bz][G3];     // 96 KB
    int tid  = threadIdx.x;
    int wave = tid >> 5;
    int lane = tid & 31;

    for (int e = tid; e < Bz * Hz; e += 1024) {
        float v = enc_hidden[e];
        sh_h[e >> 9][e & (Hz - 1)]   = v;
        sh_hbf[e >> 9][e & (Hz - 1)] = f2bf(v);
    }
    __syncthreads();

    for (int t = 0; t < Tz; ++t) {
        // hg = h @ W_hh^T : M=16, N=1536, K=512. 96 n-tiles over 32 waves (3 each).
#pragma unroll
        for (int i = 0; i < 3; ++i) {
            int nt = wave + i * 32;
            v8f c = {};
            for (int kk = 0; kk < 16; ++kk) {
                Frag a;
                {
                    int m = lane & 15, kb = (lane >> 4) * 8;
#pragma unroll
                    for (int v = 0; v < 8; ++v) {
                        int k = kk * 32 + ((v >> 2) * 16) + kb + 2 * (v & 3);
                        a.u[v] = *(const unsigned int*)&sh_hbf[m][k];
                    }
                }
                Frag b = load_b_wT(whhbf + (size_t)nt * 16 * Hz + kk * 32, Hz);
                c = wmma_bf(a, b, c);
            }
            int nl = lane & 15, mh = (lane >> 4) * 8;
#pragma unroll
            for (int r = 0; r < 8; ++r)
                sh_hg[mh + r][nt * 16 + nl] = c[r];
        }
        __syncthreads();

        // gate update (8192 elements over 1024 threads)
        const float* xgt = xg + (size_t)t * Bz * G3;
#pragma unroll
        for (int i = 0; i < 8; ++i) {
            int e = tid + i * 1024;
            int b = e >> 9, j = e & (Hz - 1);
            float xr = xgt[b * G3 + j];
            float xz = xgt[b * G3 + Hz + j];
            float xn = xgt[b * G3 + 2 * Hz + j];
            float hr = sh_hg[b][j]          + b_hh[j];
            float hz = sh_hg[b][Hz + j]     + b_hh[Hz + j];
            float hn = sh_hg[b][2 * Hz + j] + b_hh[2 * Hz + j];
            float rg = 1.f / (1.f + __expf(-(xr + hr)));
            float zg = 1.f / (1.f + __expf(-(xz + hz)));
            float ng = tanhf(xn + rg * hn);
            float hnew = (1.f - zg) * ng + zg * sh_h[b][j];
            sh_h[b][j]   = hnew;
            unsigned short hb = f2bf(hnew);
            sh_hbf[b][j] = hb;
            outbf[((size_t)t * 16 + b) * Hz + j] = hb;
        }
        __syncthreads();
    }

    for (int e = tid; e < Bz * Hz; e += 1024)
        hlast_out[e] = sh_h[e >> 9][e & (Hz - 1)];
}

// ---------------- attention: scores -> softmax -> mix, one block per batch ----------------
__global__ __launch_bounds__(256, 1)
void k_attn(const unsigned short* __restrict__ outbf,
            const unsigned short* __restrict__ encbf,
            float* __restrict__ attn_out,
            unsigned short* __restrict__ combbf) {
    __shared__ float          sc[Tz][Sz];     // 32 KB
    __shared__ unsigned short abf[Tz][Sz];    // 16 KB
    int b    = blockIdx.x;
    int tid  = threadIdx.x;
    int wave = tid >> 5;
    int lane = tid & 31;

    // scores: M=64, N=128, K=512 -> 4x8 = 32 tiles over 8 waves
#pragma unroll
    for (int i = 0; i < 4; ++i) {
        int tile = wave + i * 8;
        int mt = tile >> 3, nt = tile & 7;
        v8f c = {};
        for (int kk = 0; kk < 16; ++kk) {
            Frag a;
            {
                int m = lane & 15, kb = (lane >> 4) * 8;
                int trow = mt * 16 + m;
                const unsigned short* arow = outbf + ((size_t)trow * 16 + b) * Hz + kk * 32;
#pragma unroll
                for (int v = 0; v < 8; ++v) {
                    int k = ((v >> 2) * 16) + kb + 2 * (v & 3);
                    a.u[v] = *(const unsigned int*)(arow + k);
                }
            }
            // B[k][n] = enc[b][s=nt*16+n][kk*32+k]  (pairs contiguous along h=k)
            Frag bb = load_b_wT(encbf + ((size_t)b * Sz + nt * 16) * Hz + kk * 32, Hz);
            c = wmma_bf(a, bb, c);
        }
        int nl = lane & 15, mh = (lane >> 4) * 8;
#pragma unroll
        for (int r = 0; r < 8; ++r)
            sc[mt * 16 + mh + r][nt * 16 + nl] = c[r];
    }
    __syncthreads();

    // softmax over S: one row per thread (threads 0..63)
    if (tid < Tz) {
        int t = tid;
        float mx = -1e30f;
        for (int s = 0; s < Sz; ++s) mx = fmaxf(mx, sc[t][s]);
        float sum = 0.f;
        for (int s = 0; s < Sz; ++s) { float e = __expf(sc[t][s] - mx); sc[t][s] = e; sum += e; }
        float inv = 1.f / sum;
        for (int s = 0; s < Sz; ++s) {
            float a = sc[t][s] * inv;
            abf[t][s] = f2bf(a);
            attn_out[(size_t)b * Tz * Sz + t * Sz + s] = a;
        }
    }
    __syncthreads();

    // mix = attn @ enc : M=64, N=512, K=128 -> 4x32 = 128 tiles over 8 waves
#pragma unroll
    for (int i = 0; i < 16; ++i) {
        int tile = wave * 16 + i;
        int mt = tile >> 5, nt = tile & 31;
        v8f c = {};
#pragma unroll
        for (int kk = 0; kk < 4; ++kk) {
            Frag a;
            {
                int m = lane & 15, kb = (lane >> 4) * 8;
#pragma unroll
                for (int v = 0; v < 8; ++v) {
                    int k = kk * 32 + ((v >> 2) * 16) + kb + 2 * (v & 3);
                    a.u[v] = *(const unsigned int*)&abf[mt * 16 + m][k];
                }
            }
            Frag bb;
            {
                int n = lane & 15, kb2 = (lane >> 4) * 8;
#pragma unroll
                for (int v = 0; v < 8; ++v) {
                    int k = kk * 32 + ((v >> 2) * 16) + kb2 + 2 * (v & 3);
                    unsigned short lo = encbf[((size_t)b * Sz + k)     * Hz + nt * 16 + n];
                    unsigned short hi = encbf[((size_t)b * Sz + k + 1) * Hz + nt * 16 + n];
                    bb.u[v] = (unsigned int)lo | ((unsigned int)hi << 16);
                }
            }
            c = wmma_bf(a, bb, c);
        }
        int nl = lane & 15, mh = (lane >> 4) * 8;
#pragma unroll
        for (int r = 0; r < 8; ++r) {
            int t = mt * 16 + mh + r;
            int h = nt * 16 + nl;
            combbf[((size_t)t * 16 + b) * H2 + h] = f2bf(c[r]);   // mix half of combined
        }
    }
    // output half of combined
    for (int e = tid; e < Tz * Hz; e += 256) {
        int t = e >> 9, h = e & (Hz - 1);
        combbf[((size_t)t * 16 + b) * H2 + Hz + h] = outbf[((size_t)t * 16 + b) * Hz + h];
    }
}

// ---------------- combined GEMM + tanh: (1024x1024) x (1024x512) ----------------
__global__ void k_gemm_comb(const unsigned short* __restrict__ combbf,
                            const unsigned short* __restrict__ awbf,
                            const float* __restrict__ attn_b,
                            unsigned short* __restrict__ out2bf) {
    int gw = (blockIdx.x * blockDim.x + threadIdx.x) >> 5;   // 2048 waves: 64 x 32 tiles
    int mt = gw >> 5, nt = gw & 31;
    int lane = threadIdx.x & 31;
    v8f c = {};
#pragma unroll 4
    for (int kk = 0; kk < 32; ++kk) {
        Frag a = load_a_rm(combbf + (size_t)mt * 16 * H2 + kk * 32, H2);
        Frag b = load_b_wT(awbf + (size_t)nt * 16 * H2 + kk * 32, H2);
        c = wmma_bf(a, b, c);
    }
    int nl = lane & 15, mh = (lane >> 4) * 8;
    int n = nt * 16 + nl;
    float bias = attn_b[n];
#pragma unroll
    for (int r = 0; r < 8; ++r) {
        int m = mt * 16 + mh + r;
        out2bf[(size_t)m * Hz + n] = f2bf(tanhf(c[r] + bias));
    }
}

// ---------------- logits GEMM: (1024x512) x (512x32000), 4x B-reuse per wave ----------------
__global__ void k_gemm_logits(const unsigned short* __restrict__ out2bf,
                              const unsigned short* __restrict__ owbf,
                              const float* __restrict__ out_b,
                              float* __restrict__ logits) {
    int gw = (blockIdx.x * blockDim.x + threadIdx.x) >> 5;   // 32000 waves: 16 mq x 2000 nt
    int mq = gw / 2000, nt = gw % 2000;
    int lane = threadIdx.x & 31;
    v8f zero = {};
    v8f c0 = zero, c1 = zero, c2 = zero, c3 = zero;
    const unsigned short* wbase = owbf + (size_t)nt * 16 * Hz;
#pragma unroll 2
    for (int kk = 0; kk < 16; ++kk) {
        __builtin_prefetch(wbase + (kk + 1) * 32, 0, 1);     // global_prefetch_b8
        Frag b = load_b_wT(wbase + kk * 32, Hz);
        Frag a0 = load_a_rm(out2bf + ((size_t)(mq * 4 + 0) * 16) * Hz + kk * 32, Hz);
        c0 = wmma_bf(a0, b, c0);
        Frag a1 = load_a_rm(out2bf + ((size_t)(mq * 4 + 1) * 16) * Hz + kk * 32, Hz);
        c1 = wmma_bf(a1, b, c1);
        Frag a2 = load_a_rm(out2bf + ((size_t)(mq * 4 + 2) * 16) * Hz + kk * 32, Hz);
        c2 = wmma_bf(a2, b, c2);
        Frag a3 = load_a_rm(out2bf + ((size_t)(mq * 4 + 3) * 16) * Hz + kk * 32, Hz);
        c3 = wmma_bf(a3, b, c3);
    }
    int nl = lane & 15, mh = (lane >> 4) * 8;
    int n = nt * 16 + nl;
    float bias = out_b[n];
    v8f cs[4] = {c0, c1, c2, c3};
#pragma unroll
    for (int i = 0; i < 4; ++i) {
#pragma unroll
        for (int r = 0; r < 8; ++r) {
            int m = (mq * 4 + i) * 16 + mh + r;
            int t = m >> 4, bidx = m & 15;
            logits[((size_t)bidx * Tz + t) * Vz + n] = cs[i][r] + bias;
        }
    }
}

// ---------------- in-place log_softmax over V, one block per (b,t) row ----------------
__global__ void k_logsoftmax(float* __restrict__ logits) {
    __shared__ float red[256];
    int row = blockIdx.x;
    float* p = logits + (size_t)row * Vz;
    int tid = threadIdx.x;
    float mx = -1e30f;
    for (int v = tid; v < Vz; v += 256) mx = fmaxf(mx, p[v]);
    red[tid] = mx; __syncthreads();
    for (int s = 128; s > 0; s >>= 1) { if (tid < s) red[tid] = fmaxf(red[tid], red[tid + s]); __syncthreads(); }
    mx = red[0]; __syncthreads();
    float sum = 0.f;
    for (int v = tid; v < Vz; v += 256) sum += __expf(p[v] - mx);
    red[tid] = sum; __syncthreads();
    for (int s = 128; s > 0; s >>= 1) { if (tid < s) red[tid] += red[tid + s]; __syncthreads(); }
    float lse = mx + __logf(red[0]);
    for (int v = tid; v < Vz; v += 256) p[v] = p[v] - lse;
}

// ---------------- host ----------------
extern "C" void kernel_launch(void* const* d_in, const int* in_sizes, int n_in,
                              void* d_out, int out_size, void* d_ws, size_t ws_size,
                              hipStream_t stream) {
    const int*   inputs = (const int*)  d_in[0];
    const float* enc_h  = (const float*)d_in[1];
    const float* enc_o  = (const float*)d_in[2];
    const float* emb    = (const float*)d_in[3];
    const float* W_ih   = (const float*)d_in[4];
    const float* W_hh   = (const float*)d_in[5];
    const float* b_ih   = (const float*)d_in[6];
    const float* b_hh   = (const float*)d_in[7];
    const float* attn_W = (const float*)d_in[8];
    const float* attn_b = (const float*)d_in[9];
    const float* out_W  = (const float*)d_in[10];
    const float* out_b  = (const float*)d_in[11];

    char* ws = (char*)d_ws;
    size_t off = 0;
    auto alloc = [&](size_t bytes) {
        void* p = ws + off;
        off = (off + bytes + 255) & ~(size_t)255;
        return p;
    };
    unsigned short* wihbf  = (unsigned short*)alloc((size_t)G3 * Hz * 2);
    unsigned short* whhbf  = (unsigned short*)alloc((size_t)G3 * Hz * 2);
    unsigned short* awbf   = (unsigned short*)alloc((size_t)Hz * H2 * 2);
    unsigned short* owbf   = (unsigned short*)alloc((size_t)Vz * Hz * 2);
    unsigned short* encbf  = (unsigned short*)alloc((size_t)Bz * Sz * Hz * 2);
    unsigned short* xbf    = (unsigned short*)alloc((size_t)Mz * Hz * 2);
    float*          xg     = (float*)         alloc((size_t)Mz * G3 * 4);
    unsigned short* outbf  = (unsigned short*)alloc((size_t)Mz * Hz * 2);
    unsigned short* combbf = (unsigned short*)alloc((size_t)Mz * H2 * 2);
    unsigned short* out2bf = (unsigned short*)alloc((size_t)Mz * Hz * 2);

    float* d_logprobs = (float*)d_out;                                   // B*T*V
    float* d_hlast    = (float*)d_out + (size_t)Bz * Tz * Vz;            // B*H
    float* d_attn     = d_hlast + (size_t)Bz * Hz;                       // B*T*S

    auto cvt = [&](const float* s, unsigned short* d, int n) {
        k_f2bf<<<(n + 255) / 256, 256, 0, stream>>>(s, d, n);
    };
    cvt(W_ih,   wihbf, G3 * Hz);
    cvt(W_hh,   whhbf, G3 * Hz);
    cvt(attn_W, awbf,  Hz * H2);
    cvt(out_W,  owbf,  Vz * Hz);
    cvt(enc_o,  encbf, Bz * Sz * Hz);

    k_gather_x<<<(Mz * Hz + 255) / 256, 256, 0, stream>>>(inputs, emb, xbf);
    k_gemm_xg<<<768, 256, 0, stream>>>(xbf, wihbf, b_ih, xg);            // 6144 waves
    k_gru<<<1, 1024, 0, stream>>>(enc_h, whhbf, b_hh, xg, outbf, d_hlast);
    k_attn<<<Bz, 256, 0, stream>>>(outbf, encbf, d_attn, combbf);
    k_gemm_comb<<<256, 256, 0, stream>>>(combbf, awbf, attn_b, out2bf);  // 2048 waves
    k_gemm_logits<<<4000, 256, 0, stream>>>(out2bf, owbf, out_b, d_logprobs); // 32000 waves
    k_logsoftmax<<<Mz, 256, 0, stream>>>(d_logprobs);
}
